// FastRCNN_29858612641855
// MI455X (gfx1250) — matrix operand order repaired and checked by reading, not compiled, not run on hardware
//
#include <hip/hip_runtime.h>
#include <cstddef>
#include <cstdint>

// ---------------- problem constants ----------------
#define NR    1024      // number of ROIs
#define FCN   4096      // FC width
#define KPOOL 25088     // 512*7*7
#define CCH   512
#define IMH   50
#define IMW   50
#define ROI7  7
#define EPSG  0.001f

typedef __attribute__((ext_vector_type(16))) __bf16 v16bf;
typedef __attribute__((ext_vector_type(8)))  __bf16 v8bf;
typedef __attribute__((ext_vector_type(8)))  float  v8f;
typedef __attribute__((ext_vector_type(4)))  float  f32x4;

// ---------------- CDNA5 async Global->LDS (signature probed from clang) -----
#ifndef __has_builtin
#define __has_builtin(x) 0
#endif
#if __has_builtin(__builtin_amdgcn_global_load_async_to_lds_b128) && \
    __has_builtin(__builtin_amdgcn_s_wait_asynccnt)
#define HAVE_ASYNC_LDS 1
#else
#define HAVE_ASYNC_LDS 0
#endif

#if HAVE_ASYNC_LDS
typedef int v4i __attribute__((vector_size(16)));
typedef __attribute__((address_space(1))) v4i AS1v4i;   // global int4
typedef __attribute__((address_space(3))) v4i AS3v4i;   // LDS int4
// copy 32 contiguous bytes (2 x b128) from global to LDS, tracked by ASYNCcnt
__device__ __forceinline__ void async_cp_b128x2(const void* g, void* l) {
  __builtin_amdgcn_global_load_async_to_lds_b128((AS1v4i*)g, (AS3v4i*)l, 0, 0);
  __builtin_amdgcn_global_load_async_to_lds_b128((AS1v4i*)((const char*)g + 16),
                                                 (AS3v4i*)((char*)l + 16), 0, 0);
}
__device__ __forceinline__ void wait_async_le2() { __builtin_amdgcn_s_wait_asynccnt(2); }
__device__ __forceinline__ void wait_async_0()   { __builtin_amdgcn_s_wait_asynccnt(0); }
#else
__device__ __forceinline__ void async_cp_b128x2(const void*, void*) {}
__device__ __forceinline__ void wait_async_le2() {}
__device__ __forceinline__ void wait_async_0()   {}
#endif

__device__ __forceinline__ __bf16 f2bf(float f) {
  unsigned u = __builtin_bit_cast(unsigned, f);
  u += 0x7FFFu + ((u >> 16) & 1u);                 // round-to-nearest-even
  unsigned short h = (unsigned short)(u >> 16);
  return __builtin_bit_cast(__bf16, h);
}

// convert 16 fp32 (in 4 f32x4 regs) -> 16 bf16 into LDS (two 16B stores)
__device__ __forceinline__ void cvt16_store(const f32x4* r, __bf16* dst) {
  v8bf lo, hi;
  lo[0] = f2bf(r[0].x); lo[1] = f2bf(r[0].y); lo[2] = f2bf(r[0].z); lo[3] = f2bf(r[0].w);
  lo[4] = f2bf(r[1].x); lo[5] = f2bf(r[1].y); lo[6] = f2bf(r[1].z); lo[7] = f2bf(r[1].w);
  hi[0] = f2bf(r[2].x); hi[1] = f2bf(r[2].y); hi[2] = f2bf(r[2].z); hi[3] = f2bf(r[2].w);
  hi[4] = f2bf(r[3].x); hi[5] = f2bf(r[3].y); hi[6] = f2bf(r[3].z); hi[7] = f2bf(r[3].w);
  *(v8bf*)dst       = lo;
  *(v8bf*)(dst + 8) = hi;
}

// ---------------------------------------------------------------------------
// WMMA GEMM:  C[M,N] = scale * (A[M,K] x B) + bias,  optional ReLU.
//   BT_KMAJOR == false : B is weight-layout  B[n*K + k]   (out = A @ W^T)
//   BT_KMAJOR == true  : B is K-major        B[k*N + n]   (out = A @ B)
// A storage fp32 (reg-double-buffered, cvt to bf16) or bf16 (async-to-LDS,
// LDS-double-buffered when available). f32 WMMA accumulation.
// Block: 256 threads = 8 wave32 waves, tile 128x128, K-step 32.
// Wave grid 2x4: each wave owns 64(M) x 32(N) = 4x2 wmma tiles.
// ---------------------------------------------------------------------------
template<typename AT, bool BT_KMAJOR, bool RELU>
__global__ __launch_bounds__(256) void gemm_wmma_kernel(
    const AT* __restrict__ A, const float* __restrict__ B,
    const float* __restrict__ bias, float* __restrict__ C,
    int M, int N, int K, float scale)
{
  constexpr bool A_IS_F32 = __is_same(AT, float);
  constexpr bool ASYNC_A  = (HAVE_ASYNC_LDS != 0) && !A_IS_F32;

  __shared__ __bf16 As[2][128][32];
  __shared__ __bf16 Bs[128][32];     // Bs[n][k]

  const int t    = threadIdx.x;
  const int lane = t & 31;
  const int w    = t >> 5;
  const int wr   = w >> 2;           // 0..1
  const int wc   = w & 3;            // 0..3
  const int lm   = lane & 15;
  const int lh   = lane >> 4;
  const size_t mblk = (size_t)blockIdx.y * 128;
  const size_t nblk = (size_t)blockIdx.x * 128;

  // staging coordinates
  const int arow = t >> 1, ak0 = (t & 1) * 16;
  const AT* aptr = A + (mblk + arow) * (size_t)K + ak0;
  const int brow = t >> 1, bk0 = (t & 1) * 16;   // weight layout
  const int bkk  = t >> 3, bc0 = (t & 7) * 16;   // K-major layout

  f32x4 aR[4];
  v8bf  aRb[2];
  f32x4 bR[4];

  auto loadA = [&](int kt) {
    if constexpr (A_IS_F32) {
      const f32x4* s4 = (const f32x4*)(aptr + kt);
      aR[0] = s4[0]; aR[1] = s4[1]; aR[2] = s4[2]; aR[3] = s4[3];
    } else if constexpr (!ASYNC_A) {
      const v8bf* s = (const v8bf*)(aptr + kt);
      aRb[0] = s[0]; aRb[1] = s[1];
    }
  };
  auto loadB = [&](int kt) {
    const float* s = BT_KMAJOR ? (B + (size_t)(kt + bkk) * N + nblk + bc0)
                               : (B + (nblk + brow) * (size_t)K + kt + bk0);
    const f32x4* s4 = (const f32x4*)s;
    bR[0] = s4[0]; bR[1] = s4[1]; bR[2] = s4[2]; bR[3] = s4[3];
  };
  auto storeA = [&]() {
    __bf16* dst = &As[0][arow][ak0];
    if constexpr (A_IS_F32) {
      cvt16_store(aR, dst);
    } else if constexpr (!ASYNC_A) {
      *(v8bf*)dst = aRb[0]; *(v8bf*)(dst + 8) = aRb[1];
    }
  };
  auto storeB = [&]() {
    if constexpr (!BT_KMAJOR) {
      cvt16_store(bR, &Bs[brow][bk0]);
    } else {
#pragma unroll
      for (int q = 0; q < 4; ++q) {
        Bs[bc0 + q * 4 + 0][bkk] = f2bf(bR[q].x);
        Bs[bc0 + q * 4 + 1][bkk] = f2bf(bR[q].y);
        Bs[bc0 + q * 4 + 2][bkk] = f2bf(bR[q].z);
        Bs[bc0 + q * 4 + 3][bkk] = f2bf(bR[q].w);
      }
    }
  };

  const v8f vzero = {0.f,0.f,0.f,0.f,0.f,0.f,0.f,0.f};
  v8f acc[4][2];
#pragma unroll
  for (int mt = 0; mt < 4; ++mt)
#pragma unroll
    for (int nt = 0; nt < 2; ++nt) acc[mt][nt] = vzero;

  // prologue: kick off tile kt=0
  if constexpr (ASYNC_A) async_cp_b128x2(aptr, &As[0][arow][ak0]);
  else                   loadA(0);
  loadB(0);

  int buf = 0;
  for (int kt = 0; kt < K; kt += 32) {
    const bool more = (kt + 32) < K;
    __syncthreads();                       // all fragment reads of prev step done
    storeB();
    if constexpr (!ASYNC_A) storeA();
    if constexpr (ASYNC_A) {
      if (more) async_cp_b128x2(aptr + kt + 32, &As[buf ^ 1][arow][ak0]);
      if (more) wait_async_le2();          // allow next tile's 2 ops in flight
      else      wait_async_0();            // drain before last compute
    }
    __syncthreads();
    if (more) { loadB(kt + 32); loadA(kt + 32); }   // overlap with WMMA below

    const int ab = ASYNC_A ? buf : 0;

    // ---- fragments from LDS (ISA 16-bit A/B wave32 layouts) ----
    union FU { v16bf v; v8bf h[2]; };
    FU a[4], b[2];
#pragma unroll
    for (int mt = 0; mt < 4; ++mt) {
      const int row = wr * 64 + mt * 16 + lm;
      a[mt].h[0] = *(const v8bf*)&As[ab][row][lh * 8];
      a[mt].h[1] = *(const v8bf*)&As[ab][row][16 + lh * 8];
    }
#pragma unroll
    for (int nt = 0; nt < 2; ++nt) {
      const int col = wc * 32 + nt * 16 + lm;
      b[nt].h[0] = *(const v8bf*)&Bs[col][lh * 8];
      b[nt].h[1] = *(const v8bf*)&Bs[col][16 + lh * 8];
    }
#pragma unroll
    for (int mt = 0; mt < 4; ++mt)
#pragma unroll
      for (int nt = 0; nt < 2; ++nt)
        acc[mt][nt] = __builtin_amdgcn_wmma_f32_16x16x32_bf16(
            false, a[mt].v, false, b[nt].v, (short)0, acc[mt][nt], false, false);

    buf ^= 1;
  }

  // ---- epilogue (C/D layout: VGPR r -> row r + 8*(lane>=16), col = lane&15) ----
#pragma unroll
  for (int mt = 0; mt < 4; ++mt) {
#pragma unroll
    for (int nt = 0; nt < 2; ++nt) {
      const size_t col = nblk + wc * 32 + nt * 16 + lm;
      const float bv = bias ? bias[col] : 0.0f;
#pragma unroll
      for (int r = 0; r < 8; ++r) {
        const size_t row = mblk + wr * 64 + mt * 16 + lh * 8 + r;
        float v = acc[mt][nt][r] * scale + bv;
        if constexpr (RELU) v = fmaxf(v, 0.0f);
        C[row * (size_t)N + col] = v;
      }
    }
  }
}

template<typename AT, bool BT, bool RELU>
static void run_gemm(const AT* A, const float* B, const float* bias, float* C,
                     int M, int N, int K, float scale, hipStream_t s) {
  dim3 grid(N / 128, M / 128);
  gemm_wmma_kernel<AT, BT, RELU><<<grid, 256, 0, s>>>(A, B, bias, C, M, N, K, scale);
}

// ---------------------------------------------------------------------------
// ROI pool -> bf16 pooled features  pool[r, c*49 + i*7 + j]
// ---------------------------------------------------------------------------
__global__ __launch_bounds__(256) void roipool_kernel(
    const float* __restrict__ x, const float* __restrict__ rois,
    const int* __restrict__ bidx, __bf16* __restrict__ pool)
{
  const int r = blockIdx.x;
  __shared__ int ix[ROI7], iy[ROI7];
  if (threadIdx.x == 0) {
    const float x1 = rois[r * 4 + 1] * 0.0625f;
    const float y1 = rois[r * 4 + 0] * 0.0625f;
    const float x2 = rois[r * 4 + 3] * 0.0625f;
    const float y2 = rois[r * 4 + 2] * 0.0625f;
    for (int k = 0; k < ROI7; ++k) {
      const float g  = (k + 0.5f) / 7.0f;
      const float sx = x1 + g * (x2 - x1);
      const float sy = y1 + g * (y2 - y1);
      int jx = (int)rintf(sx); jx = jx < 0 ? 0 : (jx > IMW - 1 ? IMW - 1 : jx);
      int jy = (int)rintf(sy); jy = jy < 0 ? 0 : (jy > IMH - 1 ? IMH - 1 : jy);
      ix[k] = jx; iy[k] = jy;
    }
  }
  __syncthreads();
  const float* xb = x + (size_t)bidx[r] * CCH * IMH * IMW;
  for (int idx = threadIdx.x; idx < KPOOL; idx += 256) {
    const int c = idx / 49, rem = idx % 49, i = rem / 7, j = rem % 7;
    pool[(size_t)r * KPOOL + idx] = f2bf(xb[c * (IMH * IMW) + iy[i] * IMW + ix[j]]);
  }
}

// cx, cy, hw, hh per ROI (using xy-swapped rois like the reference)
__global__ void geom_kernel(const float* __restrict__ rois, float* __restrict__ geom) {
  const int r = blockIdx.x * blockDim.x + threadIdx.x;
  if (r >= NR) return;
  const float x1 = rois[r * 4 + 1], y1 = rois[r * 4 + 0];
  const float x2 = rois[r * 4 + 3], y2 = rois[r * 4 + 2];
  const float cx = 0.5f * (x1 + x2), cy = 0.5f * (y1 + y2);
  geom[r * 4 + 0] = cx;
  geom[r * 4 + 1] = cy;
  geom[r * 4 + 2] = fabsf(cx - x2);
  geom[r * 4 + 3] = fabsf(cy - y2);
}

// transposed gate: G[j,i] = relu(w . tg[i,j,:] + b) + 1
__global__ __launch_bounds__(256) void gate_kernel(
    const float* __restrict__ geom, const float* __restrict__ wv,
    const float* __restrict__ bv, float* __restrict__ G)
{
  const int j = blockIdx.x;
  const float cxj = geom[j * 4 + 0], cyj = geom[j * 4 + 1];
  const float hwj = geom[j * 4 + 2], hhj = geom[j * 4 + 3];
  const float w0 = wv[0], w1 = wv[1], w2 = wv[2], w3 = wv[3], bb = bv[0];
  for (int i = threadIdx.x; i < NR; i += 256) {
    const float cxi = geom[i * 4 + 0], cyi = geom[i * 4 + 1];
    const float hwi = geom[i * 4 + 2], hhi = geom[i * 4 + 3];
    const float tx = fmaxf(fabsf(cxi - cxj), EPSG) / hwi;
    const float ty = fmaxf(fabsf(cyi - cyj), EPSG) / hhi;
    const float tw = hwi / hwj;
    const float th = hhi / hhj;
    const float v = w0 * tx + w1 * ty + w2 * tw + w3 * th + bb;
    G[(size_t)j * NR + i] = fmaxf(v, 0.0f) + 1.0f;
  }
}

// rowwise: wnT[j,i] = exp(P[j,i]-max_i)*G[j,i] / sum_i(...)
__global__ __launch_bounds__(256) void softmax_gate_kernel(
    const float* __restrict__ P, const float* __restrict__ G,
    float* __restrict__ Wn, int N)
{
  __shared__ float red[256];
  const int j = blockIdx.x;
  const float* prow = P + (size_t)j * N;
  const float* grow = G + (size_t)j * N;
  float* wrow = Wn + (size_t)j * N;

  float m = -3.402823466e38f;
  for (int i = threadIdx.x; i < N; i += 256) m = fmaxf(m, prow[i]);
  red[threadIdx.x] = m; __syncthreads();
  for (int s = 128; s > 0; s >>= 1) {
    if (threadIdx.x < s) red[threadIdx.x] = fmaxf(red[threadIdx.x], red[threadIdx.x + s]);
    __syncthreads();
  }
  m = red[0]; __syncthreads();

  float sum = 0.0f;
  for (int i = threadIdx.x; i < N; i += 256) {
    const float e = expf(prow[i] - m) * grow[i];
    wrow[i] = e;
    sum += e;
  }
  red[threadIdx.x] = sum; __syncthreads();
  for (int s = 128; s > 0; s >>= 1) {
    if (threadIdx.x < s) red[threadIdx.x] += red[threadIdx.x + s];
    __syncthreads();
  }
  const float inv = 1.0f / red[0];
  for (int i = threadIdx.x; i < N; i += 256) wrow[i] *= inv;
}

__global__ void add_kernel(const float* __restrict__ a, const float* __restrict__ b,
                           float* __restrict__ c, int n) {
  const int i = blockIdx.x * blockDim.x + threadIdx.x;
  if (i < n) c[i] = a[i] + b[i];
}

// tiny-N final linears (N = 84 / 21): one thread per output, fp32 dot
__global__ __launch_bounds__(256) void linear_small_kernel(
    const float* __restrict__ A, const float* __restrict__ W,
    const float* __restrict__ b, float* __restrict__ out, int M, int N, int K)
{
  const int idx = blockIdx.x * blockDim.x + threadIdx.x;
  if (idx >= M * N) return;
  const int m = idx / N, n = idx % N;
  const float* a = A + (size_t)m * K;
  const float* w = W + (size_t)n * K;
  float s = 0.0f;
  for (int k = 0; k < K; k += 4) {
    const f32x4 av = *(const f32x4*)(a + k);
    const f32x4 wv = *(const f32x4*)(w + k);
    s += av.x * wv.x + av.y * wv.y + av.z * wv.z + av.w * wv.w;
  }
  out[idx] = s + b[n];
}

// ---------------------------------------------------------------------------
extern "C" void kernel_launch(void* const* d_in, const int* in_sizes, int n_in,
                              void* d_out, int out_size, void* d_ws, size_t ws_size,
                              hipStream_t stream)
{
  (void)in_sizes; (void)n_in; (void)out_size; (void)ws_size;

  const float* x        = (const float*)d_in[0];
  const float* rois     = (const float*)d_in[1];
  const int*   ridx     = (const int*)  d_in[2];
  const float* fc7_w    = (const float*)d_in[3];  const float* fc7_b   = (const float*)d_in[4];
  const float* fc8_w    = (const float*)d_in[5];  const float* fc8_b   = (const float*)d_in[6];
  const float* wk1_w    = (const float*)d_in[7];  const float* wk1_b   = (const float*)d_in[8];
  const float* wq1_w    = (const float*)d_in[9];  const float* wq1_b   = (const float*)d_in[10];
  const float* wv1_w    = (const float*)d_in[11]; const float* wv1_b   = (const float*)d_in[12];
  const float* wk2_w    = (const float*)d_in[13]; const float* wk2_b   = (const float*)d_in[14];
  const float* wq2_w    = (const float*)d_in[15]; const float* wq2_b   = (const float*)d_in[16];
  const float* wv2_w    = (const float*)d_in[17]; const float* wv2_b   = (const float*)d_in[18];
  const float* wg1_w    = (const float*)d_in[19]; const float* wg1_b   = (const float*)d_in[20];
  const float* wg2_w    = (const float*)d_in[21]; const float* wg2_b   = (const float*)d_in[22];
  const float* cls_w    = (const float*)d_in[23]; const float* cls_b   = (const float*)d_in[24];
  const float* loc_w    = (const float*)d_in[25]; const float* loc_b   = (const float*)d_in[26];
  const float* score_w  = (const float*)d_in[27]; const float* score_b = (const float*)d_in[28];
  float* out = (float*)d_out;

  // ---- workspace carve-up (256B aligned) ----
  char* ws = (char*)d_ws;
  size_t o = 0;
  auto carve = [&](size_t bytes) -> void* {
    void* p = ws + o;
    o += (bytes + 255) & ~(size_t)255;
    return p;
  };
  float*  G1    = (float*)carve((size_t)NR * NR * 4);
  float*  G2    = (float*)carve((size_t)NR * NR * 4);
  float*  geom  = (float*)carve((size_t)NR * 4 * 4);
  float*  fc7   = (float*)carve((size_t)NR * FCN * 4);
  float*  fc8   = (float*)carve((size_t)NR * FCN * 4);
  float*  sum   = (float*)carve((size_t)NR * FCN * 4);
  float*  wnT   = (float*)carve((size_t)NR * NR * 4);
  float*  Pm    = (float*)carve((size_t)NR * NR * 4);
  float*  fccls = (float*)carve((size_t)NR * FCN * 4);
  float*  Kb    = (float*)carve((size_t)NR * FCN * 4);
  float*  Qb    = (float*)carve((size_t)NR * FCN * 4);
  float*  Vb    = (float*)carve((size_t)NR * FCN * 4);
  float*  relb  = (float*)carve((size_t)NR * FCN * 4);
  // pool (bf16, 51MB) aliases the K/Q/V/rel region (64MB) — dead before K1 runs
  __bf16* poolbf = (__bf16*)Kb;

  // ---- pipeline ----
  roipool_kernel<<<NR, 256, 0, stream>>>(x, rois, ridx, poolbf);
  geom_kernel<<<(NR + 255) / 256, 256, 0, stream>>>(rois, geom);
  gate_kernel<<<NR, 256, 0, stream>>>(geom, wg1_w, wg1_b, G1);
  gate_kernel<<<NR, 256, 0, stream>>>(geom, wg2_w, wg2_b, G2);

  // fc7 = relu(pool @ fc7_w^T + b)
  run_gemm<__bf16, false, true >(poolbf, fc7_w, fc7_b, fc7, NR, FCN, KPOOL, 1.0f, stream);

  // ---- relation block 1 ----
  run_gemm<float, false, false>(fc7, wk1_w, wk1_b, Kb, NR, FCN, FCN, 1.0f, stream);
  run_gemm<float, false, false>(fc7, wq1_w, wq1_b, Qb, NR, FCN, FCN, 1.0f, stream);
  run_gemm<float, false, false>(fc7, wv1_w, wv1_b, Vb, NR, FCN, FCN, 1.0f, stream);
  // P[j,i] = dot(Q_j, K_i)/64  (transposed attention logits)
  run_gemm<float, false, false>(Qb, Kb, nullptr, Pm, NR, NR, FCN, 1.0f / 64.0f, stream);
  softmax_gate_kernel<<<NR, 256, 0, stream>>>(Pm, G1, wnT, NR);
  // rel[j,c] = sum_i wnT[j,i] * V[i,c]   (B is K-major)
  run_gemm<float, true, false>(wnT, Vb, nullptr, relb, NR, FCN, NR, 1.0f, stream);
  add_kernel<<<(NR * FCN) / 256, 256, 0, stream>>>(fc7, relb, sum, NR * FCN);
  run_gemm<float, false, true >(sum, fc8_w, fc8_b, fc8, NR, FCN, FCN, 1.0f, stream);

  // ---- relation block 2 ----
  run_gemm<float, false, false>(fc8, wk2_w, wk2_b, Kb, NR, FCN, FCN, 1.0f, stream);
  run_gemm<float, false, false>(fc8, wq2_w, wq2_b, Qb, NR, FCN, FCN, 1.0f, stream);
  run_gemm<float, false, false>(fc8, wv2_w, wv2_b, Vb, NR, FCN, FCN, 1.0f, stream);
  run_gemm<float, false, false>(Qb, Kb, nullptr, Pm, NR, NR, FCN, 1.0f / 64.0f, stream);
  softmax_gate_kernel<<<NR, 256, 0, stream>>>(Pm, G2, wnT, NR);
  run_gemm<float, true, false>(wnT, Vb, nullptr, relb, NR, FCN, NR, 1.0f, stream);
  add_kernel<<<(NR * FCN) / 256, 256, 0, stream>>>(fc8, relb, sum, NR * FCN);

  // fc_cls = sum @ cls_w^T + b   (no relu)
  run_gemm<float, false, false>(sum, cls_w, cls_b, fccls, NR, FCN, FCN, 1.0f, stream);

  // outputs: locs (1024x84) then scores (1024x21), concatenated flat
  linear_small_kernel<<<(NR * 84 + 255) / 256, 256, 0, stream>>>(
      fccls, loc_w, loc_b, out, NR, 84, FCN);
  linear_small_kernel<<<(NR * 21 + 255) / 256, 256, 0, stream>>>(
      fccls, score_w, score_b, out + (size_t)NR * 84, NR, 21, FCN);
}
